// VectorQuantizer_6708738917034
// MI455X (gfx1250) — matrix operand order repaired and checked by reading, not compiled, not run on hardware
//
#include <hip/hip_runtime.h>

typedef __attribute__((ext_vector_type(16))) _Float16 v16h;
typedef __attribute__((ext_vector_type(8)))  _Float16 v8h;
typedef __attribute__((ext_vector_type(8)))  float    v8f;
typedef __attribute__((ext_vector_type(4)))  unsigned int v4u;
typedef __attribute__((ext_vector_type(8)))  int      v8i;
typedef __attribute__((ext_vector_type(4)))  int      v4i;

#define C_DIM   256
#define HW      1024           // 32*32 spatial positions per batch image
#define N_ROWS  32768          // 32 * 1024 z-vectors
#define K_CODES 1024
#define ZQ_ELEMS 8388608       // 32*256*32*32
#define TM      128            // z-rows per workgroup (8 waves x 16 rows)
#define TKC     32             // codebook rows per LDS chunk (double-buffered)
#define NCHUNK  (K_CODES / TKC)
#define LDB     264            // halfs per code row in LDS (TDM pad: +8 halfs/row)

// ---------------------------------------------------------------------------
// Kernel 0: codebook f32 -> f16, contiguous [1024][256] rows in ws.
// grid=256, block=256, one float4 per thread.
// ---------------------------------------------------------------------------
__global__ __launch_bounds__(256) void vq_cb16(const float* __restrict__ cb,
                                               _Float16* __restrict__ cb16) {
  const size_t i4 = (size_t)blockIdx.x * 256 + threadIdx.x;
  float4 v = ((const float4*)cb)[i4];
  _Float16* dst = cb16 + i4 * 4;
  dst[0] = (_Float16)v.x; dst[1] = (_Float16)v.y;
  dst[2] = (_Float16)v.z; dst[3] = (_Float16)v.w;
}

// ---------------------------------------------------------------------------
// Kernel 1: exact f32 squared norms of codebook rows.  grid=1024, block=32
// ---------------------------------------------------------------------------
__global__ __launch_bounds__(32) void vq_esq(const float* __restrict__ cb,
                                             float* __restrict__ esq) {
  const int k    = blockIdx.x;
  const int lane = threadIdx.x;
  const float4* row = (const float4*)(cb + (size_t)k * C_DIM);
  float4 a = row[lane * 2 + 0];
  float4 b = row[lane * 2 + 1];
  float s = a.x*a.x + a.y*a.y + a.z*a.z + a.w*a.w
          + b.x*b.x + b.y*b.y + b.z*b.z + b.w*b.w;
#pragma unroll
  for (int off = 16; off >= 1; off >>= 1) s += __shfl_xor(s, off, 32);
  if (lane == 0) esq[k] = s;
}

// ---------------------------------------------------------------------------
// TDM descriptor: load TKC f16 codebook rows (256 halfs each) into LDS,
// padding each 128-dword row with 4 dwords -> LDS row stride = 264 halfs.
// ---------------------------------------------------------------------------
__device__ __forceinline__ void tdm_load_chunk(const _Float16* cb16,
                                               unsigned lds_off, int chunk) {
  const unsigned long long ga =
      (unsigned long long)(uintptr_t)(cb16 + (size_t)chunk * TKC * C_DIM);
  v4u g0;
  g0.x = 1u;                                   // count=1, user descriptor
  g0.y = lds_off;                              // lds_addr (bytes)
  g0.z = (unsigned)(ga & 0xFFFFFFFFu);         // global_addr[31:0]
  g0.w = (unsigned)((ga >> 32) & 0x01FFFFFFu)  // global_addr[56:32]
         | (2u << 30);                         // type = 2 ("image")
  v8i g1;
  g1[0] = (1 << 16)      // data_size = 2 bytes
        | (1 << 20)      // pad_enable
        | (6 << 22)      // pad_interval: 128 dwords (= one 256-half row)
        | (3 << 25);     // pad_amount: 4 dwords (= 8 halfs)
  g1[1] = (int)(256u << 16);    // tensor_dim0 = 256 elems/row  (bits 63:48)
  g1[2] = (int)(1024u << 16);   // tensor_dim1 = 1024 rows      (bits 95:80)
  g1[3] = (int)(256u << 16);    // tile_dim0  = 256             (bits 127:112)
  g1[4] = TKC;                  // tile_dim1  = TKC             (bits 143:128)
  g1[5] = 256;                  // tensor_dim0_stride = 256     (bits 191:160)
  g1[6] = 0;
  g1[7] = 0;
  const v4i zz4 = {0, 0, 0, 0};              // 2D tensor: groups 2/3 unused
  const v8i zz8 = {0, 0, 0, 0, 0, 0, 0, 0};  // unused trailing group (clang-23 form)
  __builtin_amdgcn_tensor_load_to_lds(g0, g1, zz4, zz4, zz8, 0);
}

// ---------------------------------------------------------------------------
// Kernel 2: WMMA f16 distance GEMM + argmin.  grid=256, block=256 (8 waves).
// A fragments (16 z-rows/wave) cached in VGPRs; codebook chunks DMA'd by the
// Tensor Data Mover into double-buffered LDS, overlapped with compute.
// dist = e_sq[k] - 2*(z.e)  (z_sq is row-constant, irrelevant to argmin).
// ---------------------------------------------------------------------------
__global__ __launch_bounds__(256) void vq_argmin(
    const float* __restrict__ z, const _Float16* __restrict__ cb16,
    const float* __restrict__ esq, int* __restrict__ idx_out,
    float* __restrict__ idxf_out) {
  __shared__ __align__(16) _Float16 cbs[2 * TKC * LDB];   // 33792 B
  __shared__ float esqs[K_CODES];                         //  4096 B

  const int tid   = threadIdx.x;
  const int wave  = tid >> 5;
  const int lane  = tid & 31;
  const int laneM = lane & 15;   // A: row in 16-row tile; B/C: column N
  const int half  = lane >> 4;

  const int n0 = blockIdx.x * TM;
  const int b  = n0 >> 10;
  const int m0 = n0 & (HW - 1);

  // stage all 1024 e_sq values once
#pragma unroll
  for (int i = 0; i < K_CODES / 256; ++i) esqs[i * 256 + tid] = esq[i * 256 + tid];

  // kick off first TDM chunk (wave 0 only; EXECZ branch skips other waves)
  if (tid < 32) tdm_load_chunk(cb16, (unsigned)(uintptr_t)(void*)&cbs[0], 0);

  // ---- A fragments: z rows (stride-HW columns) -> f16 registers, loaded once.
  const float* zp = z + (size_t)b * (C_DIM * HW) + (m0 + wave * 16 + laneM);
  v16h afrag[8];
#pragma unroll
  for (int kc8 = 0; kc8 < 8; ++kc8) {
#pragma unroll
    for (int j = 0; j < 8; ++j) {
      const int k0 = kc8 * 32 + half * 8 + j;
      afrag[kc8][j]     = (_Float16)zp[(size_t)k0 * HW];
      afrag[kc8][j + 8] = (_Float16)zp[(size_t)(k0 + 16) * HW];
    }
  }

  float bestd[8];
  int   besti[8];
#pragma unroll
  for (int v = 0; v < 8; ++v) { bestd[v] = 3.4e38f; besti[v] = 0; }

  for (int chunk = 0; chunk < NCHUNK; ++chunk) {
    if (tid < 32) __builtin_amdgcn_s_wait_tensorcnt(0);  // chunk buffer filled
    __syncthreads();   // all waves: buffer ready AND prev compute done
    if (chunk + 1 < NCHUNK && tid < 32)
      tdm_load_chunk(cb16,
                     (unsigned)(uintptr_t)(void*)&cbs[((chunk + 1) & 1) * TKC * LDB],
                     chunk + 1);

    const _Float16* buf = cbs + (chunk & 1) * (TKC * LDB);
    const _Float16* brow0 = buf + laneM * LDB;          // col tile 0
    const _Float16* brow1 = buf + (16 + laneM) * LDB;   // col tile 1

    // 4 independent accumulator chains: 2 col tiles x 2 split-K chains
    v8f a0a = {0,0,0,0,0,0,0,0}, a0b = {0,0,0,0,0,0,0,0};
    v8f a1a = {0,0,0,0,0,0,0,0}, a1b = {0,0,0,0,0,0,0,0};
#pragma unroll
    for (int kc8 = 0; kc8 < 8; kc8 += 2) {
      union { v16h v; v8h h[2]; } b00, b10, b01, b11;
      b00.h[0] = *(const v8h*)(brow0 + kc8 * 32 + half * 8);
      b00.h[1] = *(const v8h*)(brow0 + kc8 * 32 + half * 8 + 16);
      b10.h[0] = *(const v8h*)(brow1 + kc8 * 32 + half * 8);
      b10.h[1] = *(const v8h*)(brow1 + kc8 * 32 + half * 8 + 16);
      b01.h[0] = *(const v8h*)(brow0 + (kc8 + 1) * 32 + half * 8);
      b01.h[1] = *(const v8h*)(brow0 + (kc8 + 1) * 32 + half * 8 + 16);
      b11.h[0] = *(const v8h*)(brow1 + (kc8 + 1) * 32 + half * 8);
      b11.h[1] = *(const v8h*)(brow1 + (kc8 + 1) * 32 + half * 8 + 16);
      a0a = __builtin_amdgcn_wmma_f32_16x16x32_f16(false, afrag[kc8],     false, b00.v, (short)0, a0a, false, false);
      a1a = __builtin_amdgcn_wmma_f32_16x16x32_f16(false, afrag[kc8],     false, b10.v, (short)0, a1a, false, false);
      a0b = __builtin_amdgcn_wmma_f32_16x16x32_f16(false, afrag[kc8 + 1], false, b01.v, (short)0, a0b, false, false);
      a1b = __builtin_amdgcn_wmma_f32_16x16x32_f16(false, afrag[kc8 + 1], false, b11.v, (short)0, a1b, false, false);
    }

    // C/D layout: slot v -> (M = v + half*8, N = laneM)
    const float es0 = esqs[chunk * TKC + laneM];
    const float es1 = esqs[chunk * TKC + 16 + laneM];
    const int   c0  = chunk * TKC + laneM;
    const int   c1  = c0 + 16;
#pragma unroll
    for (int v = 0; v < 8; ++v) {
      const float d0 = es0 - 2.0f * (a0a[v] + a0b[v]);
      const float d1 = es1 - 2.0f * (a1a[v] + a1b[v]);
      if (d0 < bestd[v]) { bestd[v] = d0; besti[v] = c0; }  // ascending codes:
      if (d1 < bestd[v]) { bestd[v] = d1; besti[v] = c1; }  // strict < keeps lowest
    }
  }

  // min-reduce across the 16 lanes sharing each output row
#pragma unroll
  for (int v = 0; v < 8; ++v) {
    float d = bestd[v];
    int   i = besti[v];
#pragma unroll
    for (int off = 8; off >= 1; off >>= 1) {
      const float d2 = __shfl_xor(d, off, 32);
      const int   i2 = __shfl_xor(i, off, 32);
      if (d2 < d || (d2 == d && i2 < i)) { d = d2; i = i2; }
    }
    if (laneM == 0) {
      const int row = n0 + wave * 16 + half * 8 + v;
      idx_out[row]  = i;
      idxf_out[row] = (float)i;
    }
  }
}

// ---------------------------------------------------------------------------
// Kernel 3: gather z_q (exact f32 codebook) + per-block SSE partials.
// grid=8192, block=256, float4 per thread.  o = b*262144 + c*1024 + m.
// ---------------------------------------------------------------------------
__global__ __launch_bounds__(256) void vq_gather(
    const float* __restrict__ z, const float* __restrict__ cb,
    const int* __restrict__ idx, float* __restrict__ zq,
    float* __restrict__ partials) {
  __shared__ float red[8];
  const int tid = threadIdx.x;
  const size_t o = ((size_t)blockIdx.x * 256 + tid) * 4;
  const int b   = (int)(o >> 18);
  const int rem = (int)(o & 262143);
  const int c   = rem >> 10;
  const int m   = rem & 1023;
  const int4 k4 = *(const int4*)(idx + b * HW + m);
  float4 e;
  e.x = cb[(size_t)k4.x * C_DIM + c];   // 1MB codebook: L2-resident gather
  e.y = cb[(size_t)k4.y * C_DIM + c];
  e.z = cb[(size_t)k4.z * C_DIM + c];
  e.w = cb[(size_t)k4.w * C_DIM + c];
  const float4 zv = *(const float4*)(z + o);
  *(float4*)(zq + o) = e;               // z_q_st == z_q numerically
  const float dx = zv.x - e.x, dy = zv.y - e.y, dz = zv.z - e.z, dw = zv.w - e.w;
  float s = dx*dx + dy*dy + dz*dz + dw*dw;
#pragma unroll
  for (int off = 16; off >= 1; off >>= 1) s += __shfl_xor(s, off, 32);
  if ((tid & 31) == 0) red[tid >> 5] = s;
  __syncthreads();
  if (tid == 0) {
    float t = 0.f;
#pragma unroll
    for (int w = 0; w < 8; ++w) t += red[w];
    partials[blockIdx.x] = t;
  }
}

// ---------------------------------------------------------------------------
// Kernel 4: deterministic loss reduction. vq_loss = 1.25 * mean((z - z_q)^2)
// ---------------------------------------------------------------------------
__global__ __launch_bounds__(256) void vq_finalize(
    const float* __restrict__ partials, float* __restrict__ loss_out) {
  __shared__ float red[8];
  const int tid = threadIdx.x;
  float s = 0.f;
  for (int i = tid; i < ZQ_ELEMS / 1024; i += 256) s += partials[i];
#pragma unroll
  for (int off = 16; off >= 1; off >>= 1) s += __shfl_xor(s, off, 32);
  if ((tid & 31) == 0) red[tid >> 5] = s;
  __syncthreads();
  if (tid == 0) {
    float t = 0.f;
#pragma unroll
    for (int w = 0; w < 8; ++w) t += red[w];
    *loss_out = 1.25f * (t / (float)ZQ_ELEMS);
  }
}

// ---------------------------------------------------------------------------
extern "C" void kernel_launch(void* const* d_in, const int* in_sizes, int n_in,
                              void* d_out, int out_size, void* d_ws, size_t ws_size,
                              hipStream_t stream) {
  const float* z  = (const float*)d_in[0];   // [32,256,32,32]
  const float* cb = (const float*)d_in[1];   // [1024,256]

  float* out  = (float*)d_out;
  float* zq   = out;                 // 8388608 floats
  float* loss = out + ZQ_ELEMS;      // 1 float
  float* idxf = out + ZQ_ELEMS + 1;  // 32768 floats (indices)

  float*     wsf      = (float*)d_ws;
  float*     esq      = wsf;                                  // 1024 f32
  int*       idx      = (int*)(wsf + K_CODES);                // 32768 i32
  float*     partials = wsf + K_CODES + N_ROWS;               // 8192 f32
  _Float16*  cb16     = (_Float16*)(wsf + K_CODES + N_ROWS + 8192);  // 256K f16

  vq_cb16    <<<256,            256, 0, stream>>>(cb, cb16);
  vq_esq     <<<K_CODES,         32, 0, stream>>>(cb, esq);
  vq_argmin  <<<N_ROWS / TM,    256, 0, stream>>>(z, cb16, esq, idx, idxf);
  vq_gather  <<<ZQ_ELEMS / 1024, 256, 0, stream>>>(z, cb, idx, zq, partials);
  vq_finalize<<<1,              256, 0, stream>>>(partials, loss);
}
